// Attention_22127671509753
// MI455X (gfx1250) — matrix operand order repaired and checked by reading, not compiled, not run on hardware
//
#include <hip/hip_runtime.h>
#include <hip/hip_bf16.h>
#include <stdint.h>

typedef __bf16 v16bf __attribute__((ext_vector_type(16)));
typedef __bf16 v8bf  __attribute__((ext_vector_type(8)));
typedef float  v8f   __attribute__((ext_vector_type(8)));
typedef unsigned int v4u __attribute__((ext_vector_type(4)));
typedef int v4i __attribute__((ext_vector_type(4)));
typedef int v8i __attribute__((ext_vector_type(8)));

#define MDIM  2048
#define NQKV  6144
#define SEQ   2048
#define HD    128

#if defined(__AMDGCN__) && __has_builtin(__builtin_amdgcn_tensor_load_to_lds)
#define USE_TDM 1
#else
#define USE_TDM 0
#endif

#if USE_TDM
// One TDM 2-D tile load (bf16 elements) per CDNA5 D# layout (ISA 08 §8).
__device__ __forceinline__ void tdm_load_2d(unsigned lds_off, const void* gptr,
                                            unsigned tile_d0, unsigned tile_d1,
                                            unsigned stride0) {
  unsigned long long ga = (unsigned long long)(uintptr_t)gptr;
  v4u g0;
  g0.x = 1u;                                                // count=1, gather off
  g0.y = lds_off;                                           // lds_addr (bytes)
  g0.z = (unsigned)ga;                                      // global_addr[31:0]
  g0.w = ((unsigned)(ga >> 32) & 0x01FFFFFFu) | (2u << 30); // addr[56:32] | type=2
  v8i g1;
  g1[0] = (int)(1u << 16);          // data_size = 2 bytes
  g1[1] = (int)(tile_d0 << 16);     // tensor_dim0[15:0] (== tile: no OOB)
  g1[2] = (int)(tile_d1 << 16);     // tensor_dim1[15:0]
  g1[3] = (int)(tile_d0 << 16);     // tile_dim0
  g1[4] = (int)tile_d1;             // tile_dim1 (tile_dim2 = 0 -> 2D)
  g1[5] = (int)stride0;             // tensor_dim0_stride[31:0]
  g1[6] = 0;
  g1[7] = 0;
  v4i z = (v4i)0;
#if __clang_major__ >= 23
  v8i z8 = (v8i)0;
  __builtin_amdgcn_tensor_load_to_lds(g0, g1, z, z, z8, 0);
#else
  __builtin_amdgcn_tensor_load_to_lds(g0, g1, z, z, 0);
#endif
}
#endif

__device__ __forceinline__ v8f wmma_bf16(v16bf a, v16bf b, v8f c) {
  return __builtin_amdgcn_wmma_f32_16x16x32_bf16(false, a, false, b, (short)0, c,
                                                 false, false);
}

// A fragment: lane<16 -> row=lane, K={0..7,16..23}; lane>=16 -> row, K={8..15,24..31}
__device__ __forceinline__ v16bf frag_a(const __bf16* base, int stride, int lane) {
  int row = lane & 15;
  int klo = (lane >> 4) * 8;
  const __bf16* p = base + row * stride + klo;
  v8bf lo = *(const v8bf*)p;
  v8bf hi = *(const v8bf*)(p + 16);
  return __builtin_shufflevector(lo, hi, 0,1,2,3,4,5,6,7,8,9,10,11,12,13,14,15);
}

// B fragment from [n][k] row-major tile: lane<16 -> n=lane, K={0..7,16..23}; lane>=16 -> K={8..15,24..31}
__device__ __forceinline__ v16bf frag_b(const __bf16* base, int stride, int lane) {
  int n  = lane & 15;
  int kb = (lane >> 4) * 8;
  const __bf16* p = base + n * stride + kb;
  v8bf lo = *(const v8bf*)p;
  v8bf hi = *(const v8bf*)(p + 16);
  return __builtin_shufflevector(lo, hi, 0,1,2,3,4,5,6,7,8,9,10,11,12,13,14,15);
}

// ---------------------------------------------------------------- LayerNorm
__global__ __launch_bounds__(256) void ln_kernel(const float* __restrict__ x,
                                                 const float* __restrict__ w,
                                                 const float* __restrict__ b,
                                                 __bf16* __restrict__ out) {
  int row = blockIdx.x;
  const float* xr = x + (size_t)row * MDIM;
  int tid = threadIdx.x;
  float s = 0.f, s2 = 0.f;
  float v[8];
#pragma unroll
  for (int i = 0; i < 8; i++) {
    v[i] = xr[tid + i * 256];
    s += v[i];
    s2 += v[i] * v[i];
  }
  __shared__ float r0[256];
  __shared__ float r1[256];
  r0[tid] = s; r1[tid] = s2;
  __syncthreads();
  for (int off = 128; off > 0; off >>= 1) {
    if (tid < off) { r0[tid] += r0[tid + off]; r1[tid] += r1[tid + off]; }
    __syncthreads();
  }
  float mean = r0[0] * (1.f / MDIM);
  float var  = r1[0] * (1.f / MDIM) - mean * mean;
  float rstd = rsqrtf(var + 1e-5f);
#pragma unroll
  for (int i = 0; i < 8; i++) {
    int idx = tid + i * 256;
    out[(size_t)row * MDIM + idx] = (__bf16)((v[i] - mean) * rstd * w[idx] + b[idx]);
  }
}

// ---------------------------------------------------------------- f32 -> bf16
__global__ __launch_bounds__(256) void cvt_kernel(const float* __restrict__ in,
                                                  __bf16* __restrict__ out, int n) {
  int i = blockIdx.x * 256 + threadIdx.x;
  int stride = gridDim.x * 256;
  for (; i < n; i += stride) out[i] = (__bf16)in[i];
}

// ---------------------------------------------------------------- GEMM
// C[r][n] = sum_k A[r][k] * W[n][k] + bias[n]; block tile 64 rows x 256 cols,
// 8 waves of 32x64 (2x4 WMMA tiles), K-step 32, TDM double-buffered LDS staging.
template <bool F32OUT>
__global__ __launch_bounds__(256) void gemm_kernel(const __bf16* __restrict__ A,
                                                   const __bf16* __restrict__ W,
                                                   const float* __restrict__ bias,
                                                   void* __restrict__ outp,
                                                   int N, int K) {
  __shared__ __attribute__((aligned(32))) __bf16 As[2][64 * 32];
  __shared__ __attribute__((aligned(32))) __bf16 Ws[2][256 * 32];
  int tid = threadIdx.x, lane = tid & 31, wv = tid >> 5;
  int r0 = blockIdx.y * 64, c0 = blockIdx.x * 256;
  int wr = (wv & 1) * 32, wc = (wv >> 1) * 64;
  v8f acc[2][4] = {};

#if USE_TDM
  if (wv == 0) {
    tdm_load_2d((unsigned)(uintptr_t)(void*)As[0], A + (size_t)r0 * K, 32, 64, K);
    tdm_load_2d((unsigned)(uintptr_t)(void*)Ws[0], W + (size_t)c0 * K, 32, 256, K);
  }
#endif
  for (int k0 = 0; k0 < K; k0 += 32) {
#if USE_TDM
    int buf = (k0 >> 5) & 1;
    if (wv == 0) __builtin_amdgcn_s_wait_tensorcnt(0);  // current tiles landed
    __syncthreads();  // everyone past previous tiles; current tiles visible
    if (wv == 0 && k0 + 32 < K) {  // prefetch next K-slice into other buffer
      tdm_load_2d((unsigned)(uintptr_t)(void*)As[buf ^ 1],
                  A + (size_t)r0 * K + (k0 + 32), 32, 64, K);
      tdm_load_2d((unsigned)(uintptr_t)(void*)Ws[buf ^ 1],
                  W + (size_t)c0 * K + (k0 + 32), 32, 256, K);
    }
    const __bf16* as  = As[buf];
    const __bf16* wsb = Ws[buf];
#else
    __syncthreads();
    {
      int row = tid >> 2, ch = (tid & 3) * 8;
      *(uint4*)(As[0] + row * 32 + ch) =
          *(const uint4*)(A + (size_t)(r0 + row) * K + k0 + ch);
    }
#pragma unroll
    for (int i = 0; i < 4; i++) {
      int lin = tid + i * 256;
      int row = lin >> 2, ch = (lin & 3) * 8;
      *(uint4*)(Ws[0] + row * 32 + ch) =
          *(const uint4*)(W + (size_t)(c0 + row) * K + k0 + ch);
    }
    __syncthreads();
    const __bf16* as  = As[0];
    const __bf16* wsb = Ws[0];
#endif
    // issue all fragment loads first, then drain into 8 back-to-back WMMAs
    v16bf af[2], bfr[4];
#pragma unroll
    for (int t = 0; t < 2; t++) af[t] = frag_a(as + (wr + t * 16) * 32, 32, lane);
#pragma unroll
    for (int t = 0; t < 4; t++) bfr[t] = frag_b(wsb + (wc + t * 16) * 32, 32, lane);
#pragma unroll
    for (int i = 0; i < 2; i++)
#pragma unroll
      for (int j = 0; j < 4; j++) acc[i][j] = wmma_bf16(af[i], bfr[j], acc[i][j]);
  }

  int half = lane >> 4, cl = lane & 15;
#pragma unroll
  for (int j = 0; j < 4; j++) {
    int col = c0 + wc + j * 16 + cl;
    float bv = bias[col];
#pragma unroll
    for (int i = 0; i < 2; i++) {
      size_t o = (size_t)(r0 + wr + i * 16 + half * 8) * N + col;
      if constexpr (F32OUT) {
        float* out = (float*)outp;
#pragma unroll
        for (int r = 0; r < 8; r++) out[o + (size_t)r * N] = acc[i][j][r] + bv;
      } else {
        __bf16* out = (__bf16*)outp;
#pragma unroll
        for (int r = 0; r < 8; r++) out[o + (size_t)r * N] = (__bf16)(acc[i][j][r] + bv);
      }
    }
  }
}

// ---------------------------------------------------------------- Flash attention
__global__ __launch_bounds__(256) void attn_kernel(const __bf16* __restrict__ qkv,
                                                   __bf16* __restrict__ ctxout) {
  int q0  = blockIdx.x * 128;
  int h   = blockIdx.y;
  int bb  = blockIdx.z;
  int tid = threadIdx.x, lane = tid & 31, wv = tid >> 5;
  int cl = lane & 15, half = lane >> 4;

  __shared__ __attribute__((aligned(32))) __bf16 Ks[64 * 128];   // [key][d]
  __shared__ __attribute__((aligned(32))) __bf16 Vt[128 * 64];   // [d][key]
  __shared__ __attribute__((aligned(32))) __bf16 Ps[8][16 * 64]; // per-wave P

  v16bf qf[4];
  {
    int qrow = q0 + wv * 16 + cl;
    const __bf16* qp = qkv + ((size_t)(bb * SEQ + qrow)) * NQKV + h * HD;
    int klo = half * 8;
#pragma unroll
    for (int c = 0; c < 4; c++) {
      v8bf lo = *(const v8bf*)(qp + c * 32 + klo);
      v8bf hi = *(const v8bf*)(qp + c * 32 + klo + 16);
      qf[c] = __builtin_shufflevector(lo, hi, 0,1,2,3,4,5,6,7,8,9,10,11,12,13,14,15);
    }
  }

  v8f ctxa[8] = {};
  float mx[8], lsum[8];
#pragma unroll
  for (int r = 0; r < 8; r++) { mx[r] = -3.0e38f; lsum[r] = 0.f; }
  const float scale = 0.08838834764831845f;  // 1/sqrt(128)

  int nkv = q0 / 64 + 2;
  for (int kb = 0; kb < nkv; kb++) {
    int kv0 = kb * 64;
    __syncthreads();
#if USE_TDM
    if (wv == 0)
      tdm_load_2d((unsigned)(uintptr_t)(void*)Ks,
                  qkv + ((size_t)(bb * SEQ + kv0)) * NQKV + 2048 + h * HD,
                  128, 64, NQKV);
#else
#pragma unroll
    for (int i = 0; i < 4; i++) {
      int lin = tid + i * 256;
      int key = lin >> 4, ch = (lin & 15) * 8;
      *(uint4*)(Ks + key * 128 + ch) =
          *(const uint4*)(qkv + ((size_t)(bb * SEQ + kv0 + key)) * NQKV + 2048 + h * HD + ch);
    }
#endif
#pragma unroll
    for (int i = 0; i < 4; i++) {  // V transposed (TDM cannot transpose)
      int lin = tid + i * 256;
      int key = lin >> 4, ch = (lin & 15) * 8;
      v8bf vv = *(const v8bf*)(qkv + ((size_t)(bb * SEQ + kv0 + key)) * NQKV + 4096 + h * HD + ch);
#pragma unroll
      for (int j = 0; j < 8; j++) Vt[(ch + j) * 64 + key] = vv[j];
    }
#if USE_TDM
    if (wv == 0) __builtin_amdgcn_s_wait_tensorcnt(0);
#endif
    __syncthreads();

    // scores: hoist all 4 B fragments of a d-step, then 4 back-to-back WMMAs
    v8f sacc[4] = {};
#pragma unroll
    for (int ds = 0; ds < 4; ds++) {
      v16bf bk[4];
#pragma unroll
      for (int t = 0; t < 4; t++)
        bk[t] = frag_b(Ks + t * 16 * 128 + ds * 32, 128, lane);
#pragma unroll
      for (int t = 0; t < 4; t++) sacc[t] = wmma_bf16(qf[ds], bk[t], sacc[t]);
    }

    float mnew[8];
#pragma unroll
    for (int r = 0; r < 8; r++) {
      int qrow = q0 + wv * 16 + r + half * 8;
      float rm = mx[r];
#pragma unroll
      for (int t = 0; t < 4; t++) {
        int key = kv0 + t * 16 + cl;
        float s = sacc[t][r] * scale;
        s = (key <= qrow) ? s : -3.0e38f;
        sacc[t][r] = s;
        rm = fmaxf(rm, s);
      }
#pragma unroll
      for (int m = 1; m < 16; m <<= 1) rm = fmaxf(rm, __shfl_xor(rm, m, 32));
      mnew[r] = rm;
    }
#pragma unroll
    for (int r = 0; r < 8; r++) {
      float corr = __expf(mx[r] - mnew[r]);
      float ps = 0.f;
#pragma unroll
      for (int t = 0; t < 4; t++) {
        float p = __expf(sacc[t][r] - mnew[r]);
        ps += p;
        Ps[wv][(r + half * 8) * 64 + t * 16 + cl] = (__bf16)p;
      }
#pragma unroll
      for (int m = 1; m < 16; m <<= 1) ps += __shfl_xor(ps, m, 32);
      lsum[r] = lsum[r] * corr + ps;
      mx[r] = mnew[r];
#pragma unroll
      for (int dt = 0; dt < 8; dt++) ctxa[dt][r] *= corr;
    }
    __syncthreads();

    // ctx += P(16x64) x V(64x128): preload 4 V fragments per half, then WMMA burst
#pragma unroll
    for (int ks = 0; ks < 2; ks++) {
      v16bf pf = frag_a(Ps[wv] + ks * 32, 64, lane);
#pragma unroll
      for (int g = 0; g < 2; g++) {
        v16bf bv[4];
#pragma unroll
        for (int t = 0; t < 4; t++)
          bv[t] = frag_b(Vt + (g * 4 + t) * 16 * 64 + ks * 32, 64, lane);
#pragma unroll
        for (int t = 0; t < 4; t++)
          ctxa[g * 4 + t] = wmma_bf16(pf, bv[t], ctxa[g * 4 + t]);
      }
    }
  }

  size_t obase = ((size_t)(bb * SEQ + q0 + wv * 16 + half * 8)) * MDIM + h * HD + cl;
#pragma unroll
  for (int dt = 0; dt < 8; dt++)
#pragma unroll
    for (int r = 0; r < 8; r++)
      ctxout[obase + (size_t)r * MDIM + dt * 16] = (__bf16)(ctxa[dt][r] / lsum[r]);
}

// ---------------------------------------------------------------- launch
extern "C" void kernel_launch(void* const* d_in, const int* in_sizes, int n_in,
                              void* d_out, int out_size, void* d_ws, size_t ws_size,
                              hipStream_t stream) {
  (void)in_sizes; (void)n_in; (void)out_size; (void)ws_size;
  const float* x    = (const float*)d_in[0];
  const float* ln_w = (const float*)d_in[1];
  const float* ln_b = (const float*)d_in[2];
  const float* qkvw = (const float*)d_in[3];
  const float* qkvb = (const float*)d_in[4];
  const float* ow   = (const float*)d_in[5];
  const float* ob   = (const float*)d_in[6];
  float* out = (float*)d_out;

  char* ws = (char*)d_ws;
  __bf16* xn  = (__bf16*)ws; ws += (size_t)4096 * 2048 * 2;
  __bf16* wq  = (__bf16*)ws; ws += (size_t)6144 * 2048 * 2;
  __bf16* wo  = (__bf16*)ws; ws += (size_t)2048 * 2048 * 2;
  __bf16* qkv = (__bf16*)ws; ws += (size_t)4096 * 6144 * 2;
  __bf16* ctx = (__bf16*)ws;

  ln_kernel<<<4096, 256, 0, stream>>>(x, ln_w, ln_b, xn);
  cvt_kernel<<<2048, 256, 0, stream>>>(qkvw, wq, 6144 * 2048);
  cvt_kernel<<<1024, 256, 0, stream>>>(ow, wo, 2048 * 2048);
  gemm_kernel<false><<<dim3(6144 / 256, 4096 / 64), 256, 0, stream>>>(
      xn, wq, qkvb, qkv, 6144, 2048);
  attn_kernel<<<dim3(16, 16, 2), 256, 0, stream>>>(qkv, ctx);
  gemm_kernel<true><<<dim3(2048 / 256, 4096 / 64), 256, 0, stream>>>(
      ctx, wo, ob, out, 2048, 2048);
}
